// MultiHeadSelfAttention_63024350102143
// MI455X (gfx1250) — matrix operand order repaired and checked by reading, not compiled, not run on hardware
//
#include <hip/hip_runtime.h>
#include <hip/hip_bf16.h>
#include <math.h>

typedef __attribute__((ext_vector_type(16))) _Float16 v16h;
typedef __attribute__((ext_vector_type(8)))  float    v8f;

#define T_SEQ 2048
#define C_DIM 1024
#define HEADS 16
#define DH    64
#define BATCH 4
#define MTOT  (BATCH * T_SEQ)   // 8192

// ---------------------------------------------------------------------------
// CDNA5 async global->LDS copy (ASYNCcnt) via inline asm (CDNA5 ISA)
// ---------------------------------------------------------------------------
__device__ __forceinline__ void async_g2l_b128(void* lds, const void* g) {
  const unsigned           l  = (unsigned)(size_t)lds;            // LDS offset = addr[31:0]
  const unsigned long long ga = (unsigned long long)(size_t)g;
  asm volatile("global_load_async_to_lds_b128 %0, %1, off"
               :: "v"(l), "v"(ga) : "memory");
}
__device__ __forceinline__ void wait_async0() {
  asm volatile("s_wait_asynccnt 0x0" ::: "memory");
}

// ---------------------------------------------------------------------------
// WMMA fragment gathers in contiguous k order k = 16h + s, same for A and B (not AMD's table order).
// A 16x32 f16: lane&15 = M; lane-half selects K 0..15 | 16..31;
// 16 contiguous halves -> 2x b128 (works on LDS or global pointers).
// ---------------------------------------------------------------------------
__device__ __forceinline__ v16h load_A(const _Float16* tile, int stride,
                                       int row_base, int k0) {
  const int lane = threadIdx.x & 31;
  const int khi  = (lane >> 4) * 16;
  const _Float16* p = tile + (size_t)(row_base + (lane & 15)) * stride + k0 + khi;
  v16h a;
#pragma unroll
  for (int v = 0; v < 8; ++v) {
    const int kb = 2 * v;
    a[2 * v]     = p[kb];
    a[2 * v + 1] = p[kb + 1];
  }
  return a;
}

// B 32x16 f16 from a K-contiguous ("transposed") tile T[n][k]:
// lane&15 = N col, lane-half selects K 0..15 | 16..31 -> 16 contiguous halves
// = 2x ds_load_b128.
__device__ __forceinline__ v16h load_B_t(const _Float16* tile, int stride,
                                         int k0, int n0) {
  const int lane = threadIdx.x & 31;
  const int khi  = (lane >> 4) * 16;
  const _Float16* p = tile + (size_t)(n0 + (lane & 15)) * stride + k0 + khi;
  v16h b;
#pragma unroll
  for (int v = 0; v < 16; ++v) b[v] = p[v];
  return b;
}

__device__ __forceinline__ uint2 pack4(float a, float b, float c, float d) {
  union { _Float16 h[4]; uint2 u; } t;
  t.h[0] = (_Float16)a; t.h[1] = (_Float16)b;
  t.h[2] = (_Float16)c; t.h[3] = (_Float16)d;
  return t.u;
}

// ---------------------------------------------------------------------------
// Kernel 1: fused QKV projection.  X[8192,1024]f32 @ W[1024,1024]f32 + b.
// 256 threads / 8 waves; block tile 128x128, K-chunk 64; wave tile 32x64
// (16 WMMA per wave per chunk).  Output f16 in [B,H,T,Dh]; Q pre-scaled.
// ---------------------------------------------------------------------------
__global__ __launch_bounds__(256)
void qkv_proj_kernel(const float* __restrict__ X,
                     const float* __restrict__ Wq, const float* __restrict__ bq,
                     const float* __restrict__ Wk, const float* __restrict__ bk,
                     const float* __restrict__ Wv, const float* __restrict__ bv,
                     _Float16* __restrict__ Qh, _Float16* __restrict__ Kh,
                     _Float16* __restrict__ Vh) {
  const float* W; const float* bias; _Float16* dst; float oscale;
  if (blockIdx.z == 0)      { W = Wq; bias = bq; dst = Qh; oscale = 0.125f; }
  else if (blockIdx.z == 1) { W = Wk; bias = bk; dst = Kh; oscale = 1.0f; }
  else                      { W = Wv; bias = bv; dst = Vh; oscale = 1.0f; }

  __shared__ __align__(16) _Float16 Xs[128][72];  // [m][k], row-major
  __shared__ __align__(16) _Float16 Wt[128][72];  // [n][k], K-contiguous

  const int tid  = threadIdx.x;
  const int lane = tid & 31;
  const int wv   = tid >> 5;          // 0..7
  const int wm   = (wv >> 1) * 32;    // 0,32,64,96
  const int wn   = (wv & 1)  * 64;    // 0,64
  const int m0   = blockIdx.y * 128;
  const int n0   = blockIdx.x * 128;

  v8f acc[2][4];
#pragma unroll
  for (int i = 0; i < 2; ++i)
#pragma unroll
    for (int j = 0; j < 4; ++j)
#pragma unroll
      for (int v = 0; v < 8; ++v) acc[i][j][v] = 0.0f;

  for (int k0 = 0; k0 < C_DIM; k0 += 64) {
    // X tile 128x64 f32->f16 (float4 loads, b64 LDS stores)
#pragma unroll
    for (int p = 0; p < 8; ++p) {
      const int e = p * 256 + tid;
      const int r = e >> 4, c4 = (e & 15) * 4;
      const float4 f = *(const float4*)&X[(size_t)(m0 + r) * C_DIM + k0 + c4];
      *(uint2*)&Xs[r][c4] = pack4(f.x, f.y, f.z, f.w);
    }
    // W tile 64(k)x128(n) -> transposed Wt[n][k] (coalesced b32 loads, b64 stores)
#pragma unroll
    for (int p = 0; p < 8; ++p) {
      const int e = p * 256 + tid;
      const int n = e & 127, kq = (e >> 7) * 4;
      const size_t g = (size_t)(k0 + kq) * C_DIM + n0 + n;
      *(uint2*)&Wt[n][kq] = pack4(W[g], W[g + C_DIM], W[g + 2 * C_DIM], W[g + 3 * C_DIM]);
    }
    __syncthreads();

#pragma unroll
    for (int kc = 0; kc < 2; ++kc) {
      const v16h a0 = load_A(&Xs[0][0], 72, wm,      kc * 32);
      const v16h a1 = load_A(&Xs[0][0], 72, wm + 16, kc * 32);
#pragma unroll
      for (int j = 0; j < 4; ++j) {
        const v16h bf = load_B_t(&Wt[0][0], 72, kc * 32, wn + j * 16);
        acc[0][j] = __builtin_amdgcn_wmma_f32_16x16x32_f16(
            false, a0, false, bf, (short)0, acc[0][j], false, false);
        acc[1][j] = __builtin_amdgcn_wmma_f32_16x16x32_f16(
            false, a1, false, bf, (short)0, acc[1][j], false, false);
      }
    }
    __syncthreads();
  }

  const int hi = lane >> 4;
#pragma unroll
  for (int i = 0; i < 2; ++i)
#pragma unroll
    for (int j = 0; j < 4; ++j) {
      const int n = n0 + wn + j * 16 + (lane & 15);
      const float bval = bias[n];
      const int h = n >> 6, d = n & 63;
#pragma unroll
      for (int v = 0; v < 8; ++v) {
        const int m = m0 + wm + i * 16 + v + 8 * hi;
        const int bb = m >> 11, t = m & (T_SEQ - 1);
        dst[((size_t)(bb * HEADS + h) * T_SEQ + t) * DH + d] =
            (_Float16)((acc[i][j][v] + bval) * oscale);
      }
    }
}

// ---------------------------------------------------------------------------
// Kernel 2: flash attention.  256 threads / 8 waves; 128 queries per block,
// key tiles of 64.  K tile loaded with global_load_async_to_lds_b128;
// V stored transposed (Vt[d][key]) so P*V B-fragments are K-contiguous.
// Q A-fragments load straight from global (b128 pairs).
// ---------------------------------------------------------------------------
__global__ __launch_bounds__(256)
void attn_kernel(const _Float16* __restrict__ Qh, const _Float16* __restrict__ Kh,
                 const _Float16* __restrict__ Vh, _Float16* __restrict__ Ah) {
  __shared__ __align__(16) _Float16 Ks[64][72];      // [key][d]
  __shared__ __align__(16) _Float16 Vt[64][72];      // [d][key]  (transposed)
  __shared__ __align__(16) _Float16 Ps[8][16][72];   // per-wave P scratch

  const int tid  = threadIdx.x;
  const int lane = tid & 31;
  const int wv   = tid >> 5;      // 0..7, wave owns 16 query rows
  const int hi   = lane >> 4;
  const int bh   = blockIdx.y;
  const int q0   = blockIdx.x * 128;

  const _Float16* Qb = Qh + (size_t)bh * T_SEQ * DH;
  const _Float16* Kb = Kh + (size_t)bh * T_SEQ * DH;
  const _Float16* Vb = Vh + (size_t)bh * T_SEQ * DH;

  // Q fragments directly from global memory (row-major [t][d], stride 64).
  v16h qa[2];
#pragma unroll
  for (int i = 0; i < 2; ++i)
    qa[i] = load_A(Qb + (size_t)q0 * DH, DH, wv * 16, i * 32);

  v8f o[4];
  float m[8], l[8];
#pragma unroll
  for (int d = 0; d < 4; ++d)
#pragma unroll
    for (int v = 0; v < 8; ++v) o[d][v] = 0.0f;
#pragma unroll
  for (int v = 0; v < 8; ++v) { m[v] = -3.0e38f; l[v] = 0.0f; }

  for (int kt = 0; kt < T_SEQ / 64; ++kt) {
    const int k0 = kt * 64;
    __syncthreads();
    // K tile: async DMA global->LDS, 512 b128 chunks / 256 threads.
#pragma unroll
    for (int p = 0; p < 2; ++p) {
      const int e = p * 256 + tid;
      const int r = e >> 3, c8 = (e & 7) * 8;
      async_g2l_b128(&Ks[r][c8], &Kb[(size_t)(k0 + r) * DH + c8]);
    }
    // V tile: load b64, scatter-transpose into Vt[d][key].
#pragma unroll
    for (int p = 0; p < 4; ++p) {
      const int e = p * 256 + tid;
      const int key = e >> 4, d4 = (e & 15) * 4;
      const ushort4 q = *(const ushort4*)&Vb[(size_t)(k0 + key) * DH + d4];
      *(unsigned short*)&Vt[d4 + 0][key] = q.x;
      *(unsigned short*)&Vt[d4 + 1][key] = q.y;
      *(unsigned short*)&Vt[d4 + 2][key] = q.z;
      *(unsigned short*)&Vt[d4 + 3][key] = q.w;
    }
    if (kt + 1 < T_SEQ / 64) {
      const int r = tid & 63;
      __builtin_prefetch(&Kb[(size_t)(k0 + 64 + r) * DH], 0, 0);
      __builtin_prefetch(&Vb[(size_t)(k0 + 64 + r) * DH], 0, 0);
    }
    wait_async0();
    __syncthreads();

    // S = Q*K^T (scale folded into Q).  B = K^T from Ks[key][d]: n=key row.
    v8f s[4];
#pragma unroll
    for (int nk = 0; nk < 4; ++nk) {
#pragma unroll
      for (int v = 0; v < 8; ++v) s[nk][v] = 0.0f;
#pragma unroll
      for (int kc = 0; kc < 2; ++kc) {
        const v16h bf = load_B_t(&Ks[0][0], 72, kc * 32, nk * 16);
        s[nk] = __builtin_amdgcn_wmma_f32_16x16x32_f16(
            false, qa[kc], false, bf, (short)0, s[nk], false, false);
      }
    }

    // Online softmax: C/D rows live in VGPR index, 16 lanes/half share a row.
#pragma unroll
    for (int v = 0; v < 8; ++v) {
      float t = fmaxf(fmaxf(s[0][v], s[1][v]), fmaxf(s[2][v], s[3][v]));
#pragma unroll
      for (int off = 1; off < 16; off <<= 1)
        t = fmaxf(t, __shfl_xor(t, off, 32));
      const float mn    = fmaxf(m[v], t);
      const float alpha = __expf(m[v] - mn);
      l[v] *= alpha;
#pragma unroll
      for (int d = 0; d < 4; ++d) o[d][v] *= alpha;
      m[v] = mn;

      float rs = 0.0f;
#pragma unroll
      for (int nk = 0; nk < 4; ++nk) {
        const float pv = __expf(s[nk][v] - mn);
        rs += pv;
        Ps[wv][v + 8 * hi][nk * 16 + (lane & 15)] = (_Float16)pv;
      }
#pragma unroll
      for (int off = 1; off < 16; off <<= 1)
        rs += __shfl_xor(rs, off, 32);
      l[v] += rs;
    }

    // Re-read P in A-fragment layout (wave-local LDS; DS ops in order).
    v16h pa[2];
#pragma unroll
    for (int i = 0; i < 2; ++i)
      pa[i] = load_A(&Ps[wv][0][0], 72, 0, i * 32);

    // O += P*V.  B from Vt[d][key]: n=d row, k=key contiguous.
#pragma unroll
    for (int nd = 0; nd < 4; ++nd)
#pragma unroll
      for (int kc = 0; kc < 2; ++kc) {
        const v16h bf = load_B_t(&Vt[0][0], 72, kc * 32, nd * 16);
        o[nd] = __builtin_amdgcn_wmma_f32_16x16x32_f16(
            false, pa[kc], false, bf, (short)0, o[nd], false, false);
      }
  }

  // Epilogue: O/l -> attn[B*T, C] f16 for the output projection.
  const int b = bh >> 4, h = bh & 15;
#pragma unroll
  for (int v = 0; v < 8; ++v) {
    const float inv = 1.0f / l[v];
    const int t = q0 + wv * 16 + v + 8 * hi;
    const size_t row = (size_t)b * T_SEQ + t;
#pragma unroll
    for (int nd = 0; nd < 4; ++nd) {
      const int col = h * DH + nd * 16 + (lane & 15);
      Ah[row * C_DIM + col] = (_Float16)(o[nd][v] * inv);
    }
  }
}

// ---------------------------------------------------------------------------
// Kernel 3: output projection. attn[8192,1024]f16 @ Wo + bo -> out f32.
// Same 128x128 tiling; A tile via async DMA, Wo transposed in LDS.
// ---------------------------------------------------------------------------
__global__ __launch_bounds__(256)
void out_proj_kernel(const _Float16* __restrict__ Ain, const float* __restrict__ W,
                     const float* __restrict__ bias, float* __restrict__ Out) {
  __shared__ __align__(16) _Float16 As[128][72];
  __shared__ __align__(16) _Float16 Wt[128][72];

  const int tid  = threadIdx.x;
  const int lane = tid & 31;
  const int wv   = tid >> 5;
  const int wm   = (wv >> 1) * 32;
  const int wn   = (wv & 1)  * 64;
  const int m0   = blockIdx.y * 128;
  const int n0   = blockIdx.x * 128;

  v8f acc[2][4];
#pragma unroll
  for (int i = 0; i < 2; ++i)
#pragma unroll
    for (int j = 0; j < 4; ++j)
#pragma unroll
      for (int v = 0; v < 8; ++v) acc[i][j][v] = 0.0f;

  for (int k0 = 0; k0 < C_DIM; k0 += 64) {
    // A tile 128x64 f16: async DMA, 1024 b128 chunks / 256 threads.
#pragma unroll
    for (int p = 0; p < 4; ++p) {
      const int e = p * 256 + tid;
      const int r = e >> 3, c8 = (e & 7) * 8;
      async_g2l_b128(&As[r][c8], &Ain[(size_t)(m0 + r) * C_DIM + k0 + c8]);
    }
#pragma unroll
    for (int p = 0; p < 8; ++p) {
      const int e = p * 256 + tid;
      const int n = e & 127, kq = (e >> 7) * 4;
      const size_t g = (size_t)(k0 + kq) * C_DIM + n0 + n;
      *(uint2*)&Wt[n][kq] = pack4(W[g], W[g + C_DIM], W[g + 2 * C_DIM], W[g + 3 * C_DIM]);
    }
    wait_async0();
    __syncthreads();

#pragma unroll
    for (int kc = 0; kc < 2; ++kc) {
      const v16h a0 = load_A(&As[0][0], 72, wm,      kc * 32);
      const v16h a1 = load_A(&As[0][0], 72, wm + 16, kc * 32);
#pragma unroll
      for (int j = 0; j < 4; ++j) {
        const v16h bf = load_B_t(&Wt[0][0], 72, kc * 32, wn + j * 16);
        acc[0][j] = __builtin_amdgcn_wmma_f32_16x16x32_f16(
            false, a0, false, bf, (short)0, acc[0][j], false, false);
        acc[1][j] = __builtin_amdgcn_wmma_f32_16x16x32_f16(
            false, a1, false, bf, (short)0, acc[1][j], false, false);
      }
    }
    __syncthreads();
  }

  const int hi = lane >> 4;
#pragma unroll
  for (int i = 0; i < 2; ++i)
#pragma unroll
    for (int j = 0; j < 4; ++j) {
      const int n = n0 + wn + j * 16 + (lane & 15);
      const float bval = bias[n];
#pragma unroll
      for (int v = 0; v < 8; ++v) {
        const int mrow = m0 + wm + i * 16 + v + 8 * hi;
        Out[(size_t)mrow * C_DIM + n] = acc[i][j][v] + bval;
      }
    }
}

// ---------------------------------------------------------------------------
// Launch.  Workspace (needs >= 64 MB):
//   [0,16MB) Q f16 [B,H,T,Dh] (pre-scaled 1/8) | [16,32MB) K | [32,48MB) V
//   [48,64MB) attn f16 [B*T, C]
// ---------------------------------------------------------------------------
extern "C" void kernel_launch(void* const* d_in, const int* in_sizes, int n_in,
                              void* d_out, int out_size, void* d_ws, size_t ws_size,
                              hipStream_t stream) {
  const float* x  = (const float*)d_in[0];
  const float* Wq = (const float*)d_in[1];
  const float* bq = (const float*)d_in[2];
  const float* Wk = (const float*)d_in[3];
  const float* bk = (const float*)d_in[4];
  const float* Wv = (const float*)d_in[5];
  const float* bv = (const float*)d_in[6];
  const float* Wo = (const float*)d_in[7];
  const float* bo = (const float*)d_in[8];

  char* ws = (char*)d_ws;
  const size_t MB = 1024u * 1024u;
  _Float16* Qh = (_Float16*)(ws + 0 * MB);
  _Float16* Kh = (_Float16*)(ws + 16 * MB);
  _Float16* Vh = (_Float16*)(ws + 32 * MB);
  _Float16* Ah = (_Float16*)(ws + 48 * MB);

  const dim3 blk(256);
  const dim3 g1(C_DIM / 128, MTOT / 128, 3);  // 8 x 64 x 3
  qkv_proj_kernel<<<g1, blk, 0, stream>>>(x, Wq, bq, Wk, bk, Wv, bv, Qh, Kh, Vh);

  const dim3 g2(T_SEQ / 128, BATCH * HEADS);  // 16 x 64
  attn_kernel<<<g2, blk, 0, stream>>>(Qh, Kh, Vh, Ah);

  const dim3 g3(C_DIM / 128, MTOT / 128);     // 8 x 64
  out_proj_kernel<<<g3, blk, 0, stream>>>(Ah, Wo, bo, (float*)d_out);
}
